// Model_81260781240810
// MI455X (gfx1250) — matrix operand order repaired and checked by reading, not compiled, not run on hardware
//
#include <hip/hip_runtime.h>

// ---------------- model dims (compile-time, from reference) ----------------
constexpr int H_ = 512;    // hidden
constexpr int V_ = 64;     // vocab
constexpr int B_ = 32;     // batch
constexpr int L_ = 2048;   // encoder length
constexpr int T_ = 32;     // decode steps
constexpr int G_ = 4 * H_; // 2048 gate width
constexpr int K2_ = 2 * H_; // 1024 concat K

typedef unsigned short u16;
typedef unsigned int   u32;
typedef __attribute__((ext_vector_type(16))) __bf16 v16bf;
typedef __attribute__((ext_vector_type(8)))  float  v8f;

union FragBF { v16bf v; u32 u[8]; };
union FragF  { v8f   v; float f[8]; };

__device__ __forceinline__ u16 f2bf(float f) {
    u32 u = __float_as_uint(f);
    u32 r = u + 0x7FFFu + ((u >> 16) & 1u);   // round-to-nearest-even
    return (u16)(r >> 16);
}
__device__ __forceinline__ float bf2f(u16 s) { return __uint_as_float((u32)s << 16); }
__device__ __forceinline__ float sigm(float x) { return 1.0f / (1.0f + __expf(-x)); }

// ---- WMMA fragment loaders (CDNA5 16x16x32 bf16 layouts, wave32) ----
// A (16x32, row-major, lane r<16 holds M=r K[0..7]+K[16..23]; lane r+16 holds K[8..15]+K[24..31])
__device__ __forceinline__ void loadA_bf(FragBF& a, const u16* A, int lda, int m0, int k0, int lane) {
    int r = lane & 15, hh = lane >> 4;
    const u32* p0 = (const u32*)(A + (m0 + r) * lda + k0 + 8 * hh);
    const u32* p1 = (const u32*)(A + (m0 + r) * lda + k0 + 16 + 8 * hh);
#pragma unroll
    for (int i = 0; i < 4; ++i) { a.u[i] = p0[i]; a.u[4 + i] = p1[i]; }
}
// B for GEMM x @ W^T: B[k][n] = W[n][k], W row-major (N,K).
// Lane l reads W[n0 + (l&15)][k0 + 8*(l>>4) .. +7] and W[n0 + (l&15)][k0 + 16 + 8*(l>>4) .. +7]
__device__ __forceinline__ void loadB_w(FragBF& b, const u16* W, int ldw, int n0, int k0, int lane) {
    int c = lane & 15, hh = lane >> 4;
    const u32* p0 = (const u32*)(W + (n0 + c) * ldw + k0 + 8 * hh);
    const u32* p1 = (const u32*)(W + (n0 + c) * ldw + k0 + 16 + 8 * hh);
#pragma unroll
    for (int i = 0; i < 4; ++i) { b.u[i] = p0[i]; b.u[4 + i] = p1[i]; }
}
__device__ __forceinline__ v8f wmma_bf(const FragBF& a, const FragBF& b, v8f c) {
    return __builtin_amdgcn_wmma_f32_16x16x32_bf16(false, a.v, false, b.v, (short)0, c, false, false);
}
__device__ __forceinline__ void zero8(FragF& f) {
#pragma unroll
    for (int i = 0; i < 8; ++i) f.f[i] = 0.f;
}

// virtual-A loaders (concat of two sources along K; chunks of 8 never straddle k=512)
__device__ __forceinline__ void load8_gates(u32* dst, int row, int kk,
                                            const u16* embbf, const int* dec_in, const float* h) {
    if (kk < H_) {
        const u32* p = (const u32*)(embbf + dec_in[row] * H_ + kk);
#pragma unroll
        for (int i = 0; i < 4; ++i) dst[i] = p[i];
    } else {
        const float* p = h + row * H_ + (kk - H_);
#pragma unroll
        for (int i = 0; i < 4; ++i)
            dst[i] = (u32)f2bf(p[2 * i]) | ((u32)f2bf(p[2 * i + 1]) << 16);
    }
}
__device__ __forceinline__ void load8_cat(u32* dst, int row, int kk,
                                          const u16* hbf, const float* ctx) {
    if (kk < H_) {
        const u32* p = (const u32*)(hbf + row * H_ + kk);
#pragma unroll
        for (int i = 0; i < 4; ++i) dst[i] = p[i];
    } else {
        const float* p = ctx + row * H_ + (kk - H_);
#pragma unroll
        for (int i = 0; i < 4; ++i)
            dst[i] = (u32)f2bf(p[2 * i]) | ((u32)f2bf(p[2 * i + 1]) << 16);
    }
}

// ---------------- one-time pack / init kernels ----------------
__global__ void k_cvt_bf16(u16* dst, const float* src, int n) {
    for (int i = blockIdx.x * blockDim.x + threadIdx.x; i < n; i += gridDim.x * blockDim.x)
        dst[i] = f2bf(src[i]);
}
// Wcat[n][k] = k<512 ? W_ih[n][k] : W_hh[n][k-512]   (2048 x 1024)
__global__ void k_pack_wcat(u16* dst, const float* wih, const float* whh) {
    const int n = G_ * K2_;
    for (int i = blockIdx.x * blockDim.x + threadIdx.x; i < n; i += gridDim.x * blockDim.x) {
        int row = i >> 10, k = i & 1023;
        float v = (k < H_) ? wih[row * H_ + k] : whh[row * H_ + (k - H_)];
        dst[i] = f2bf(v);
    }
}
// value_bf[b][l][h] = bf16(enc[l][b][h])   (transpose + convert)
__global__ void k_pack_value(u16* dst, const float* enc) {
    const int n = B_ * L_ * H_;
    for (int i = blockIdx.x * blockDim.x + threadIdx.x; i < n; i += gridDim.x * blockDim.x) {
        int b = i / (L_ * H_);
        int rem = i - b * (L_ * H_);
        int l = rem / H_, hh = rem - l * H_;
        dst[i] = f2bf(enc[(l * B_ + b) * H_ + hh]);
    }
}
__global__ void k_init(float* attn, float* hbuf, float* cbuf, int* dec_in, float* loss,
                       const float* h0, const float* c0, const int* init_dec) {
    int i = blockIdx.x * blockDim.x + threadIdx.x;
    if (i < B_ * L_) attn[i] = 0.f;
    if (i < B_ * H_) { hbuf[i] = h0[i]; cbuf[i] = c0[i]; }
    if (i < B_) dec_in[i] = init_dec[i];
    if (i == 0) loss[0] = 0.f;
}

// ---------------- v_proj = value @ Wv^T  (65536 x 512 x 512, bf16 WMMA) ----------------
// block = 256 thr = 8 waves; block owns 16 rows x 512 cols; wave w owns N range [64w, 64w+64)
__global__ void k_vproj(u16* vproj, const u16* value, const u16* wv) {
    int lane = threadIdx.x & 31, wave = threadIdx.x >> 5;
    int m0 = blockIdx.x * 16;
    int n0 = wave * 64;
    FragF acc[4];
#pragma unroll
    for (int j = 0; j < 4; ++j) zero8(acc[j]);
    for (int k0 = 0; k0 < H_; k0 += 32) {
        FragBF a; loadA_bf(a, value, H_, m0, k0, lane);
#pragma unroll
        for (int j = 0; j < 4; ++j) {
            FragBF b; loadB_w(b, wv, H_, n0 + 16 * j, k0, lane);
            acc[j].v = wmma_bf(a, b, acc[j].v);
        }
    }
    int c = lane & 15, hh = lane >> 4;
#pragma unroll
    for (int j = 0; j < 4; ++j)
#pragma unroll
        for (int r = 0; r < 8; ++r)
            vproj[(m0 + r + 8 * hh) * H_ + n0 + 16 * j + c] = f2bf(acc[j].f[r]);
}

// ---------------- per-step kernels ----------------
// gates = [emb(dec_in) | h] @ Wcat^T   (32 x 1024 x 2048) -> fp32
__global__ void k_gates(float* gates, const u16* embbf, const int* dec_in,
                        const float* hbuf, const u16* wcat) {
    int lane = threadIdx.x & 31, wave = threadIdx.x >> 5;
    int tile = blockIdx.x * 8 + wave;          // 0..255
    int m0 = (tile & 1) * 16;
    int n0 = (tile >> 1) * 16;
    int r = lane & 15, hh = lane >> 4;
    int row = m0 + r;
    FragF acc; zero8(acc);
    for (int k0 = 0; k0 < K2_; k0 += 32) {
        FragBF a;
        load8_gates(&a.u[0], row, k0 + 8 * hh,      embbf, dec_in, hbuf);
        load8_gates(&a.u[4], row, k0 + 16 + 8 * hh, embbf, dec_in, hbuf);
        FragBF b; loadB_w(b, wcat, K2_, n0, k0, lane);
        acc.v = wmma_bf(a, b, acc.v);
    }
    int c = lane & 15;
#pragma unroll
    for (int r2 = 0; r2 < 8; ++r2)
        gates[(m0 + r2 + 8 * hh) * G_ + n0 + c] = acc.f[r2];
}

// LSTM cell elementwise (torch gate order i,f,g,o)
__global__ void k_lstm(float* hbuf, float* cbuf, u16* hbf, const float* gates,
                       const float* b_ih, const float* b_hh) {
    int i = blockIdx.x * blockDim.x + threadIdx.x;
    if (i >= B_ * H_) return;
    int b = i >> 9, j = i & (H_ - 1);
    const float* g = gates + b * G_;
    float gi = g[j]            + b_ih[j]            + b_hh[j];
    float gf = g[H_ + j]       + b_ih[H_ + j]       + b_hh[H_ + j];
    float gg = g[2 * H_ + j]   + b_ih[2 * H_ + j]   + b_hh[2 * H_ + j];
    float go = g[3 * H_ + j]   + b_ih[3 * H_ + j]   + b_hh[3 * H_ + j];
    float cn = sigm(gf) * cbuf[i] + sigm(gi) * tanhf(gg);
    float hn = sigm(go) * tanhf(cn);
    cbuf[i] = cn;
    hbuf[i] = hn;
    hbf[i] = f2bf(hn);
}

// q = h_new @ Wq^T (32x512x512); also zero sum_sig for this step
__global__ void k_q(float* q, float* sum_sig, const u16* hbf, const u16* wq) {
    if (blockIdx.x == 0 && threadIdx.x < B_) sum_sig[threadIdx.x] = 0.f;
    int lane = threadIdx.x & 31, wave = threadIdx.x >> 5;
    int tile = blockIdx.x * 8 + wave;          // 0..63
    int m0 = (tile & 1) * 16;
    int n0 = (tile >> 1) * 16;
    FragF acc; zero8(acc);
    for (int k0 = 0; k0 < H_; k0 += 32) {
        FragBF a; loadA_bf(a, hbf, H_, m0, k0, lane);
        FragBF b; loadB_w(b, wq, H_, n0, k0, lane);
        acc.v = wmma_bf(a, b, acc.v);
    }
    int c = lane & 15, hh = lane >> 4;
#pragma unroll
    for (int r = 0; r < 8; ++r)
        q[(m0 + r + 8 * hh) * H_ + n0 + c] = acc.f[r];
}

// score[b,l] = sigmoid( Ws . tanh(q + v_proj + conv(attn) + bias) + bs ); sum over l
// grid: 32 b x 8 chunks; block 256 thr = 8 waves; wave = one l-row per iter (32 iters)
__global__ void k_score(float* sigb, float* sum_sig,
                        const float* q, const u16* vproj, const float* attn,
                        const float* conv_w, const float* conv_b, const float* attn_bias,
                        const float* Ws, const float* bs) {
    __shared__ float s_q[H_], s_cw0[H_], s_cw1[H_], s_cw2[H_], s_cbb[H_], s_ws[H_];
    __shared__ float s_sum[8];
    int b = blockIdx.x >> 3, chunk = blockIdx.x & 7;
    for (int i = threadIdx.x; i < H_; i += blockDim.x) {
        s_q[i]   = q[b * H_ + i];
        s_cw0[i] = conv_w[i * 3 + 0];
        s_cw1[i] = conv_w[i * 3 + 1];
        s_cw2[i] = conv_w[i * 3 + 2];
        s_cbb[i] = conv_b[i] + attn_bias[i];
        s_ws[i]  = Ws[i];
    }
    __syncthreads();
    int lane = threadIdx.x & 31, wave = threadIdx.x >> 5;
    const float bs0 = bs[0];
    const int baseL = b * L_;
    float wsum = 0.f;
    for (int it = 0; it < 32; ++it) {
        int l = chunk * 256 + wave * 32 + it;
        float am = (l > 0)      ? attn[baseL + l - 1] : 0.f;  // zero-padded conv
        float a0 =                attn[baseL + l];
        float ap = (l < L_ - 1) ? attn[baseL + l + 1] : 0.f;
        const u32* vp = (const u32*)(vproj + (baseL + l) * H_ + lane * 16);
        float acc = 0.f;
#pragma unroll
        for (int i = 0; i < 8; ++i) {
            u32 u = vp[i];
            int h0 = lane * 16 + 2 * i;
            float v0 = bf2f((u16)(u & 0xFFFFu));
            float v1 = bf2f((u16)(u >> 16));
            float t0 = s_q[h0]     + v0 + s_cw0[h0]     * am + s_cw1[h0]     * a0 + s_cw2[h0]     * ap + s_cbb[h0];
            float t1 = s_q[h0 + 1] + v1 + s_cw0[h0 + 1] * am + s_cw1[h0 + 1] * a0 + s_cw2[h0 + 1] * ap + s_cbb[h0 + 1];
            acc += s_ws[h0] * tanhf(t0) + s_ws[h0 + 1] * tanhf(t1);
        }
#pragma unroll
        for (int off = 16; off > 0; off >>= 1) acc += __shfl_xor(acc, off, 32);
        if (lane == 0) {
            float sg = sigm(acc + bs0);
            sigb[baseL + l] = sg;
            wsum += sg;
        }
    }
    if (lane == 0) s_sum[wave] = wsum;
    __syncthreads();
    if (threadIdx.x == 0) {
        float tot = 0.f;
        for (int w = 0; w < 8; ++w) tot += s_sum[w];
        atomicAdd(&sum_sig[b], tot);
    }
}

__global__ void k_norm(float* attn, const float* sigb, const float* sum_sig) {
    int i = blockIdx.x * blockDim.x + threadIdx.x;
    if (i < B_ * L_) attn[i] = sigb[i] / sum_sig[i >> 11];
}

// context[b,h] = sum_l attn[b,l] * value[b,l,h]   (block per b; thread owns 2 h)
__global__ void k_context(float* ctx, const float* attn, const u16* value) {
    int b = blockIdx.x, t = threadIdx.x;
    const u16* vb = value + b * L_ * H_;
    const float* ab = attn + b * L_;
    float a1 = 0.f, a2 = 0.f;
    for (int l = 0; l < L_; ++l) {
        float aw = ab[l];
        const u16* row = vb + l * H_;
        if (l + 8 < L_) __builtin_prefetch((const void*)(vb + (l + 8) * H_ + t), 0, 1);
        a1 += aw * bf2f(row[t]);
        a2 += aw * bf2f(row[t + 256]);
    }
    ctx[b * H_ + t] = a1;
    ctx[b * H_ + t + 256] = a2;
}

// concat_out = tanh([h | context] @ Wcc^T + b_concat)  (32 x 1024 x 512) -> bf16
__global__ void k_concat(u16* cobf, const u16* hbf, const float* ctx,
                         const u16* wcc, const float* b_concat) {
    int lane = threadIdx.x & 31, wave = threadIdx.x >> 5;
    int tile = blockIdx.x * 8 + wave;          // 0..63
    int m0 = (tile & 1) * 16;
    int n0 = (tile >> 1) * 16;
    int r = lane & 15, hh = lane >> 4;
    int row = m0 + r;
    FragF acc; zero8(acc);
    for (int k0 = 0; k0 < K2_; k0 += 32) {
        FragBF a;
        load8_cat(&a.u[0], row, k0 + 8 * hh,      hbf, ctx);
        load8_cat(&a.u[4], row, k0 + 16 + 8 * hh, hbf, ctx);
        FragBF b; loadB_w(b, wcc, K2_, n0, k0, lane);
        acc.v = wmma_bf(a, b, acc.v);
    }
    int c = lane & 15;
#pragma unroll
    for (int r2 = 0; r2 < 8; ++r2) {
        int m = m0 + r2 + 8 * hh, n = n0 + c;
        cobf[m * H_ + n] = f2bf(tanhf(acc.f[r2] + b_concat[n]));
    }
}

// logits = concat_out @ W_out^T + b_out; softmax; argmax->dec_in & preds; masked NLL
__global__ void k_out(float* dout, int* dec_in, float* loss,
                      const u16* cobf, const u16* wout, const float* b_out,
                      const int* labels, const unsigned char* mask, int t) {
    __shared__ float s_log[B_][V_];
    __shared__ float s_ce[B_], s_m[B_];
    int lane = threadIdx.x & 31, wave = threadIdx.x >> 5;   // 8 waves = 2x4 tiles
    int m0 = (wave & 1) * 16;
    int n0 = (wave >> 1) * 16;
    FragF acc; zero8(acc);
    for (int k0 = 0; k0 < H_; k0 += 32) {
        FragBF a; loadA_bf(a, cobf, H_, m0, k0, lane);
        FragBF b; loadB_w(b, wout, H_, n0, k0, lane);
        acc.v = wmma_bf(a, b, acc.v);
    }
    int c = lane & 15, hh = lane >> 4;
#pragma unroll
    for (int r = 0; r < 8; ++r)
        s_log[m0 + r + 8 * hh][n0 + c] = acc.f[r] + b_out[n0 + c];
    __syncthreads();
    if (threadIdx.x < B_) {
        int b = threadIdx.x;
        float mx = s_log[b][0]; int am = 0;
        for (int j = 1; j < V_; ++j) { float v = s_log[b][j]; if (v > mx) { mx = v; am = j; } }
        float se = 0.f;
        for (int j = 0; j < V_; ++j) se += __expf(s_log[b][j] - mx);
        int lab = labels[t * B_ + b];
        float ce = -(s_log[b][lab] - mx - __logf(se));
        float m = mask[t * B_ + b] ? 1.f : 0.f;
        s_ce[b] = ce * m;
        s_m[b]  = m;
        dec_in[b] = am;                         // greedy feedback
        dout[1 + b * T_ + t] = (float)am;       // preds.T -> (B, T)
    }
    __syncthreads();
    if (threadIdx.x == 0) {
        float sc = 0.f, sm = 0.f;
        for (int b = 0; b < B_; ++b) { sc += s_ce[b]; sm += s_m[b]; }
        float sl = (sm > 0.f) ? sc / fmaxf(sm, 1.f) : 0.f;
        float nl = loss[0] + sl;
        loss[0] = nl;
        if (t == T_ - 1) dout[0] = nl;
    }
}

// ---------------- host launcher ----------------
extern "C" void kernel_launch(void* const* d_in, const int* in_sizes, int n_in,
                              void* d_out, int out_size, void* d_ws, size_t ws_size,
                              hipStream_t stream) {
    (void)in_sizes; (void)n_in; (void)out_size; (void)ws_size;
    const int*   init_dec  = (const int*)d_in[0];
    const float* h0        = (const float*)d_in[1];
    const float* c0        = (const float*)d_in[2];
    const float* enc       = (const float*)d_in[3];
    const int*   labels    = (const int*)d_in[4];
    const unsigned char* mask = (const unsigned char*)d_in[5];
    // d_in[6] = target_length (== T_, compile-time)
    const float* embed     = (const float*)d_in[7];
    const float* W_ih      = (const float*)d_in[8];
    const float* W_hh      = (const float*)d_in[9];
    const float* b_ih      = (const float*)d_in[10];
    const float* b_hh      = (const float*)d_in[11];
    const float* W_concat  = (const float*)d_in[12];
    const float* b_concat  = (const float*)d_in[13];
    const float* W_out     = (const float*)d_in[14];
    const float* b_out     = (const float*)d_in[15];
    const float* conv_w    = (const float*)d_in[16];
    const float* conv_b    = (const float*)d_in[17];
    const float* Wq        = (const float*)d_in[18];
    const float* Wv        = (const float*)d_in[19];
    const float* Ws        = (const float*)d_in[20];
    const float* bs        = (const float*)d_in[21];
    const float* attn_bias = (const float*)d_in[22];
    float* out = (float*)d_out;

    // workspace carve-out (all offsets 256B aligned)
    char* p = (char*)d_ws;
    auto carve = [&](size_t bytes) { char* r = p; p += (bytes + 255) & ~(size_t)255; return r; };
    u16*   value_bf = (u16*)  carve((size_t)B_ * L_ * H_ * 2);   // 64 MiB
    u16*   vproj_bf = (u16*)  carve((size_t)B_ * L_ * H_ * 2);   // 64 MiB
    u16*   wcat_bf  = (u16*)  carve((size_t)G_ * K2_ * 2);
    u16*   wq_bf    = (u16*)  carve((size_t)H_ * H_ * 2);
    u16*   wcc_bf   = (u16*)  carve((size_t)H_ * K2_ * 2);
    u16*   wout_bf  = (u16*)  carve((size_t)V_ * H_ * 2);
    u16*   emb_bf   = (u16*)  carve((size_t)V_ * H_ * 2);
    u16*   wv_bf    = (u16*)  carve((size_t)H_ * H_ * 2);
    float* hbuf     = (float*)carve((size_t)B_ * H_ * 4);
    float* cbuf     = (float*)carve((size_t)B_ * H_ * 4);
    u16*   hbf      = (u16*)  carve((size_t)B_ * H_ * 2);
    float* gates    = (float*)carve((size_t)B_ * G_ * 4);
    float* q        = (float*)carve((size_t)B_ * H_ * 4);
    float* attn     = (float*)carve((size_t)B_ * L_ * 4);
    float* sigb     = (float*)carve((size_t)B_ * L_ * 4);
    float* sum_sig  = (float*)carve(256);
    float* ctx      = (float*)carve((size_t)B_ * H_ * 4);
    u16*   cobf     = (u16*)  carve((size_t)B_ * H_ * 2);
    int*   dec_in   = (int*)  carve(256);
    float* loss     = (float*)carve(256);

    // one-time pack + init + v_proj GEMM
    k_cvt_bf16 <<<256,  256, 0, stream>>>(wq_bf,   Wq,       H_ * H_);
    k_cvt_bf16 <<<256,  256, 0, stream>>>(wcc_bf,  W_concat, H_ * K2_);
    k_cvt_bf16 <<<64,   256, 0, stream>>>(wout_bf, W_out,    V_ * H_);
    k_cvt_bf16 <<<64,   256, 0, stream>>>(emb_bf,  embed,    V_ * H_);
    k_cvt_bf16 <<<256,  256, 0, stream>>>(wv_bf,   Wv,       H_ * H_);
    k_pack_wcat <<<512, 256, 0, stream>>>(wcat_bf, W_ih, W_hh);
    k_pack_value<<<2048,256, 0, stream>>>(value_bf, enc);
    k_init      <<<256, 256, 0, stream>>>(attn, hbuf, cbuf, dec_in, loss, h0, c0, init_dec);
    k_vproj     <<<(B_ * L_) / 16, 256, 0, stream>>>(vproj_bf, value_bf, wv_bf);

    // sequential decode (state carried in ws; 8 kernels / step)
    for (int t = 0; t < T_; ++t) {
        k_gates  <<<32,  256, 0, stream>>>(gates, emb_bf, dec_in, hbuf, wcat_bf);
        k_lstm   <<<64,  256, 0, stream>>>(hbuf, cbuf, hbf, gates, b_ih, b_hh);
        k_q      <<<8,   256, 0, stream>>>(q, sum_sig, hbf, wq_bf);
        k_score  <<<256, 256, 0, stream>>>(sigb, sum_sig, q, vproj_bf, attn,
                                           conv_w, conv_b, attn_bias, Ws, bs);
        k_norm   <<<256, 256, 0, stream>>>(attn, sigb, sum_sig);
        k_context<<<32,  256, 0, stream>>>(ctx, attn, value_bf);
        k_concat <<<8,   256, 0, stream>>>(cobf, hbf, ctx, wcc_bf, b_concat);
        k_out    <<<1,   256, 0, stream>>>(out, dec_in, loss, cobf, wout_bf, b_out,
                                           labels, mask, t);
    }
}
